// PointNetPPEncoder_82033875354290
// MI455X (gfx1250) — compile-verified, matrix operand order checked by reading, not run on hardware
//
#include <hip/hip_runtime.h>
#include <hip/hip_bf16.h>

typedef __attribute__((ext_vector_type(16))) _Float16 v16h;
typedef __attribute__((ext_vector_type(8)))  _Float16 v8h;
typedef __attribute__((ext_vector_type(8)))  float    v8f;

#define BN_EPS 1e-5f

// ---------------------------------------------------------------------------
// prep: build cur0 (B,N,3), feats0 (B,N,6); emit l_xyz[0] (B,3,N), l_points[0] (B,6,N)
// xyz layout: (B, 6, N)
// ---------------------------------------------------------------------------
__global__ void prep0_kernel(const float* __restrict__ xyz,
                             float* __restrict__ cur0, float* __restrict__ feats0,
                             float* __restrict__ out_lxyz0, float* __restrict__ out_lp0,
                             int B, int N)
{
    int i = blockIdx.x * blockDim.x + threadIdx.x;
    if (i >= B * N) return;
    int b = i / N, n = i % N;
#pragma unroll
    for (int c = 0; c < 6; ++c) {
        float v = xyz[((size_t)b * 6 + c) * N + n];
        feats0[((size_t)b * N + n) * 6 + c] = v;
        out_lp0[((size_t)b * 6 + c) * N + n] = v;
        if (c < 3) {
            cur0[((size_t)b * N + n) * 3 + c] = v;
            out_lxyz0[((size_t)b * 3 + c) * N + n] = v;
        }
    }
}

// ---------------------------------------------------------------------------
// Farthest point sampling: one block per batch. cur: (B, Nin, 3). fidx: (B, S)
// ---------------------------------------------------------------------------
__global__ void fps_kernel(const float* __restrict__ cur, int* __restrict__ fidx,
                           int Nin, int S)
{
    __shared__ float dist[4096];
    __shared__ float rmax[256];
    __shared__ int   rimax[256];
    int b = blockIdx.x;
    int tid = threadIdx.x;
    for (int n = tid; n < Nin; n += 256) dist[n] = 1e10f;
    __syncthreads();
    int far = 0;
    for (int it = 0; it < S; ++it) {
        if (tid == 0) fidx[b * S + it] = far;
        float cx = cur[((size_t)b * Nin + far) * 3 + 0];
        float cy = cur[((size_t)b * Nin + far) * 3 + 1];
        float cz = cur[((size_t)b * Nin + far) * 3 + 2];
        float lm = -1.0f; int li = 0;
        for (int n = tid; n < Nin; n += 256) {
            float dx = cur[((size_t)b * Nin + n) * 3 + 0] - cx;
            float dy = cur[((size_t)b * Nin + n) * 3 + 1] - cy;
            float dz = cur[((size_t)b * Nin + n) * 3 + 2] - cz;
            float d = dx * dx + dy * dy + dz * dz;
            float dn = fminf(dist[n], d);
            dist[n] = dn;
            if (dn > lm) { lm = dn; li = n; }
        }
        rmax[tid] = lm; rimax[tid] = li;
        __syncthreads();
        for (int off = 128; off > 0; off >>= 1) {
            if (tid < off) {
                bool take = (rmax[tid + off] > rmax[tid]) ||
                            (rmax[tid + off] == rmax[tid] && rimax[tid + off] < rimax[tid]);
                if (take) { rmax[tid] = rmax[tid + off]; rimax[tid] = rimax[tid + off]; }
            }
            __syncthreads();
        }
        far = rimax[0];
        __syncthreads();
    }
}

// ---------------------------------------------------------------------------
// Ball query + grouping. One wave (32 lanes) per (b,s).
// Picks K=32 lowest-index in-radius neighbors (center always qualifies),
// pads with the first. Writes new_xyz (B,S,3) and f16 A rows (M x Kpad):
// row = [dx,dy,dz, feats[0..Cf-1], 0-pad]
// ---------------------------------------------------------------------------
__global__ void group_kernel(const float* __restrict__ cur, const float* __restrict__ feats,
                             const int* __restrict__ fidx, float* __restrict__ new_xyz,
                             _Float16* __restrict__ A, float r2, int Nin, int S,
                             int Cf, int Kpad)
{
    __shared__ int nbr[32];
    int g = blockIdx.x;            // b*S + s
    int b = g / S;
    int lane = threadIdx.x;
    int fid = fidx[g];
    float cx = cur[((size_t)b * Nin + fid) * 3 + 0];
    float cy = cur[((size_t)b * Nin + fid) * 3 + 1];
    float cz = cur[((size_t)b * Nin + fid) * 3 + 2];
    if (lane == 0) {
        new_xyz[(size_t)g * 3 + 0] = cx;
        new_xyz[(size_t)g * 3 + 1] = cy;
        new_xyz[(size_t)g * 3 + 2] = cz;
    }
    int cnt = 0;
    for (int base = 0; base < Nin; base += 32) {
        int n = base + lane;
        bool inr = false;
        if (n < Nin) {
            float dx = cur[((size_t)b * Nin + n) * 3 + 0] - cx;
            float dy = cur[((size_t)b * Nin + n) * 3 + 1] - cy;
            float dz = cur[((size_t)b * Nin + n) * 3 + 2] - cz;
            inr = (dx * dx + dy * dy + dz * dz <= r2);
        }
        unsigned mask = (unsigned)__ballot(inr);
        int pos = cnt + __popc(mask & ((1u << lane) - 1u));
        if (inr && pos < 32) nbr[pos] = n;
        cnt += __popc(mask);
        if (cnt >= 32) break;
    }
    __syncthreads();
    if (lane >= cnt) nbr[lane] = nbr[0];   // cnt >= 1 always (center in radius)
    __syncthreads();
    int j = nbr[lane];
    _Float16* row = A + ((size_t)g * 32 + lane) * Kpad;
    row[0] = (_Float16)(cur[((size_t)b * Nin + j) * 3 + 0] - cx);
    row[1] = (_Float16)(cur[((size_t)b * Nin + j) * 3 + 1] - cy);
    row[2] = (_Float16)(cur[((size_t)b * Nin + j) * 3 + 2] - cz);
    for (int c = 0; c < Cf; ++c)
        row[3 + c] = (_Float16)feats[((size_t)b * Nin + j) * Cf + c];
    for (int c = 3 + Cf; c < Kpad; ++c)
        row[c] = (_Float16)0.f;
}

// ---------------------------------------------------------------------------
// Pack weights: W (Cout x Cin) f32 -> f16 padded (Cout x Kpad); fold BN into
// per-channel scale/bias: y = x*scale + bias, scale = g*rsqrt(v+eps),
// bias = (b - m)*scale + beta
// ---------------------------------------------------------------------------
__global__ void pack_weights_kernel(const float* __restrict__ W, const float* __restrict__ bb,
                                    const float* __restrict__ gamma, const float* __restrict__ beta,
                                    const float* __restrict__ mean, const float* __restrict__ var,
                                    _Float16* __restrict__ Wp, float* __restrict__ scale,
                                    float* __restrict__ bias, int Cout, int Cin, int Kpad)
{
    int i = blockIdx.x * blockDim.x + threadIdx.x;
    if (i < Cout) {
        float sc = gamma[i] * rsqrtf(var[i] + BN_EPS);
        scale[i] = sc;
        bias[i] = (bb[i] - mean[i]) * sc + beta[i];
    }
    if (i < Cout * Kpad) {
        int o = i / Kpad, c = i % Kpad;
        Wp[i] = (c < Cin) ? (_Float16)W[(size_t)o * Cin + c] : (_Float16)0.f;
    }
}

// ---------------------------------------------------------------------------
// WMMA fragment load: row-major 16-bit matrix, ISA 16x32 A layout (B mirrors
// it with N as the "row"). lane<16: halves [kc..kc+7],[kc+16..kc+23];
// lane>=16: +8. Two 16B loads per lane.
// ---------------------------------------------------------------------------
__device__ inline v16h load_frag16(const _Float16* __restrict__ base, int ld,
                                   int row, int kc, int lane)
{
    const _Float16* p = base + (size_t)row * ld + kc + ((lane & 16) ? 8 : 0);
    v8h lo = *(const v8h*)p;
    v8h hi = *(const v8h*)(p + 16);
    v16h f;
#pragma unroll
    for (int i = 0; i < 8; ++i) { f[i] = lo[i]; f[i + 8] = hi[i]; }
    return f;
}

__device__ inline void emit_val(float val, int m, int ch, const float* scale,
                                const float* bias, _Float16* outH, float* outMax,
                                int Cout, int fuse_max)
{
    val = val * scale[ch] + bias[ch];
    val = fmaxf(val, 0.f);
    if (fuse_max) {
        int grp = m >> 5;   // K = 32 rows per group
        atomicMax((int*)&outMax[(size_t)grp * Cout + ch], __float_as_int(val));
    } else {
        outH[(size_t)m * Cout + ch] = (_Float16)val;
    }
}

// ---------------------------------------------------------------------------
// GEMM + BN + ReLU (+ optional fused max-over-K via atomics).
// One wave per (MT*16 x NT*16) output tile; MT*NT WMMAs per K-step with
// MT+NT fragment loads (register blocking for wmma:vmem ratio).
// A: (M x Kp) f16 row-major. Wp: (Cout x Kp) f16 row-major (acts as B tiles).
// ---------------------------------------------------------------------------
template<int MT, int NT>
__global__ void gemm_bn_relu_t(const _Float16* __restrict__ A, const _Float16* __restrict__ Wp,
                               const float* __restrict__ scale, const float* __restrict__ bias,
                               _Float16* __restrict__ outH, float* __restrict__ outMax,
                               int M, int Kp, int Cout, int fuse_max)
{
    int lane = threadIdx.x;
    int m0 = blockIdx.x * (MT * 16);
    int n0 = blockIdx.y * (NT * 16);
    int r = lane & 15;
    v8f acc[MT][NT];
#pragma unroll
    for (int i = 0; i < MT; ++i)
#pragma unroll
        for (int j = 0; j < NT; ++j)
            acc[i][j] = (v8f){};
    for (int kc = 0; kc < Kp; kc += 32) {
        v16h af[MT], bf[NT];
#pragma unroll
        for (int i = 0; i < MT; ++i)
            af[i] = load_frag16(A, Kp, m0 + i * 16 + r, kc, lane);
#pragma unroll
        for (int j = 0; j < NT; ++j)
            bf[j] = load_frag16(Wp, Kp, n0 + j * 16 + r, kc, lane);
#pragma unroll
        for (int i = 0; i < MT; ++i)
#pragma unroll
            for (int j = 0; j < NT; ++j)
                acc[i][j] = __builtin_amdgcn_wmma_f32_16x16x32_f16(
                    false, af[i], false, bf[j], (short)0, acc[i][j], false, false);
    }
    int mh = (lane & 16) ? 8 : 0;   // C/D layout: lanes 16-31 hold M rows +8
#pragma unroll
    for (int i = 0; i < MT; ++i)
#pragma unroll
        for (int j = 0; j < NT; ++j)
#pragma unroll
            for (int v = 0; v < 8; ++v)
                emit_val(acc[i][j][v], m0 + i * 16 + mh + v, n0 + j * 16 + r,
                         scale, bias, outH, outMax, Cout, fuse_max);
}

// new_xyz (B,S,3) -> out (B,3,S)
__global__ void lxyz_out_kernel(const float* __restrict__ nxt, float* __restrict__ out,
                                int B, int S)
{
    int i = blockIdx.x * blockDim.x + threadIdx.x;
    if (i >= B * S) return;
    int b = i / S, s = i % S;
#pragma unroll
    for (int c = 0; c < 3; ++c)
        out[((size_t)b * 3 + c) * S + s] = nxt[((size_t)i) * 3 + c];
}

// feats (B,S,C) -> out (B,C,S)
__global__ void lpoints_out_kernel(const float* __restrict__ feats, float* __restrict__ out,
                                   int B, int S, int C)
{
    int i = blockIdx.x * blockDim.x + threadIdx.x;
    if (i >= B * S * C) return;
    int c = i % C;
    int g = i / C;
    int b = g / S, s = g % S;
    out[((size_t)b * C + c) * S + s] = feats[(size_t)g * C + c];
}

static inline void launch_gemm(const _Float16* A, const _Float16* Wp,
                               const float* sc, const float* bs,
                               _Float16* outH, float* outMax,
                               int M, int Kp, int Cout, int fuse_max, hipStream_t stream)
{
    if (Cout >= 64) {
        // 32x64 tiles: 8 WMMAs / 6 fragment loads per K-step
        gemm_bn_relu_t<2, 4><<<dim3(M / 32, Cout / 64), 32, 0, stream>>>(
            A, Wp, sc, bs, outH, outMax, M, Kp, Cout, fuse_max);
    } else {
        // Cout == 32: 64x32 tiles
        gemm_bn_relu_t<4, 2><<<dim3(M / 64, 1), 32, 0, stream>>>(
            A, Wp, sc, bs, outH, outMax, M, Kp, Cout, fuse_max);
    }
}

// ---------------------------------------------------------------------------
extern "C" void kernel_launch(void* const* d_in, const int* in_sizes, int n_in,
                              void* d_out, int out_size, void* d_ws, size_t ws_size,
                              hipStream_t stream)
{
    (void)in_sizes; (void)n_in; (void)out_size; (void)ws_size;
    const int B = 16, N0 = 4096, K = 32;
    const int Nin[4]  = {4096, 1024, 512, 256};
    const int Ss[4]   = {1024, 512, 256, 128};
    const int Cf[4]   = {6, 64, 128, 256};
    const float r2a[4] = {0.05f * 0.05f, 0.1f * 0.1f, 0.2f * 0.2f, 0.3f * 0.3f};
    const int mlp[4][3] = {{32, 32, 64}, {64, 64, 128}, {128, 128, 256}, {256, 256, 512}};

    const float* xyz = (const float*)d_in[0];
    float* out = (float*)d_out;
    char* ws = (char*)d_ws;

    // ---- workspace carve-up ----
    size_t cursor = 0;
    auto walloc = [&](size_t bytes) {
        size_t o = cursor;
        cursor = (cursor + bytes + 255) & ~(size_t)255;
        return o;
    };
    size_t curb[2];
    curb[0] = walloc((size_t)B * N0 * 3 * 4);
    curb[1] = walloc((size_t)B * N0 * 3 * 4);
    size_t fbytes = (size_t)B * 1024 * 64 * 4;        // >= B*4096*6*4
    if ((size_t)B * 4096 * 6 * 4 > fbytes) fbytes = (size_t)B * 4096 * 6 * 4;
    size_t featb[2];
    featb[0] = walloc(fbytes);
    featb[1] = walloc(fbytes);
    size_t fidx_o = walloc((size_t)B * 1024 * 4);
    size_t bufA_o = walloc((size_t)27 * 1024 * 1024 * 2);  // 54 MB, covers M*Kpad max
    size_t bufB_o = walloc((size_t)18 * 1024 * 1024 * 2);  // 36 MB
    size_t wp_o[12], sc_o[12], bs_o[12];
    int cin_s[12], kp_s[12], co_s[12];
    for (int L = 0; L < 4; ++L)
        for (int j = 0; j < 3; ++j) {
            int st = L * 3 + j;
            int cin = (j == 0) ? (3 + Cf[L]) : mlp[L][j - 1];
            int kp = ((cin + 31) / 32) * 32;
            int co = mlp[L][j];
            cin_s[st] = cin; kp_s[st] = kp; co_s[st] = co;
            wp_o[st] = walloc((size_t)co * kp * 2);
            sc_o[st] = walloc((size_t)co * 4);
            bs_o[st] = walloc((size_t)co * 4);
        }

    // ---- output offsets (tuple(l_xyz) then tuple(l_points), flat) ----
    const int Sx[5] = {4096, 1024, 512, 256, 128};
    size_t oxz[5];
    oxz[0] = 0;
    for (int i = 0; i < 4; ++i) oxz[i + 1] = oxz[i] + (size_t)B * 3 * Sx[i];
    size_t olp[5];
    olp[0] = oxz[4] + (size_t)B * 3 * Sx[4];
    const size_t lpsz[5] = {(size_t)6 * 4096, (size_t)64 * 1024, (size_t)128 * 512,
                            (size_t)256 * 256, (size_t)512 * 128};
    for (int i = 0; i < 4; ++i) olp[i + 1] = olp[i] + (size_t)B * lpsz[i];

    // ---- pack weights (12 stages) ----
    for (int L = 0; L < 4; ++L) {
        int base = 1 + L * 18;
        for (int j = 0; j < 3; ++j) {
            int st = L * 3 + j;
            int total = co_s[st] * kp_s[st];
            pack_weights_kernel<<<(total + 255) / 256, 256, 0, stream>>>(
                (const float*)d_in[base + j],        // W
                (const float*)d_in[base + 3 + j],    // b
                (const float*)d_in[base + 6 + j],    // gamma
                (const float*)d_in[base + 9 + j],    // beta
                (const float*)d_in[base + 12 + j],   // mean
                (const float*)d_in[base + 15 + j],   // var
                (_Float16*)(ws + wp_o[st]), (float*)(ws + sc_o[st]), (float*)(ws + bs_o[st]),
                co_s[st], cin_s[st], kp_s[st]);
        }
    }

    // ---- prep layer-0 inputs + first two outputs ----
    prep0_kernel<<<(B * N0 + 255) / 256, 256, 0, stream>>>(
        xyz, (float*)(ws + curb[0]), (float*)(ws + featb[0]),
        out + oxz[0], out + olp[0], B, N0);

    // ---- 4 set-abstraction layers ----
    int curIn = 0, featIn = 0;
    for (int L = 0; L < 4; ++L) {
        int nin = Nin[L], S = Ss[L], cf = Cf[L];
        int kp0 = kp_s[L * 3 + 0];
        int c0 = mlp[L][0], c1 = mlp[L][1], c2 = mlp[L][2];
        int M = B * S * K;
        float* cur = (float*)(ws + curb[curIn]);
        float* nxt = (float*)(ws + curb[curIn ^ 1]);
        float* fin = (float*)(ws + featb[featIn]);
        float* fout = (float*)(ws + featb[featIn ^ 1]);
        int* fidx = (int*)(ws + fidx_o);
        _Float16* Abuf = (_Float16*)(ws + bufA_o);
        _Float16* Bbuf = (_Float16*)(ws + bufB_o);

        fps_kernel<<<B, 256, 0, stream>>>(cur, fidx, nin, S);
        group_kernel<<<B * S, 32, 0, stream>>>(cur, fin, fidx, nxt, Abuf,
                                               r2a[L], nin, S, cf, kp0);
        lxyz_out_kernel<<<(B * S + 255) / 256, 256, 0, stream>>>(nxt, out + oxz[L + 1], B, S);

        // stage 0: Abuf(M,kp0) x W0 -> Bbuf(M,c0)
        launch_gemm(Abuf, (_Float16*)(ws + wp_o[L * 3 + 0]),
                    (float*)(ws + sc_o[L * 3 + 0]), (float*)(ws + bs_o[L * 3 + 0]),
                    Bbuf, nullptr, M, kp0, c0, 0, stream);
        // stage 1: Bbuf(M,c0) x W1 -> Abuf(M,c1)
        launch_gemm(Bbuf, (_Float16*)(ws + wp_o[L * 3 + 1]),
                    (float*)(ws + sc_o[L * 3 + 1]), (float*)(ws + bs_o[L * 3 + 1]),
                    Abuf, nullptr, M, c0, c1, 0, stream);
        // stage 2: Abuf(M,c1) x W2 -> fused max-over-K into fout (B*S, c2)
        hipMemsetAsync(fout, 0, (size_t)B * S * c2 * 4, stream);
        launch_gemm(Abuf, (_Float16*)(ws + wp_o[L * 3 + 2]),
                    (float*)(ws + sc_o[L * 3 + 2]), (float*)(ws + bs_o[L * 3 + 2]),
                    nullptr, fout, M, c1, c2, 1, stream);

        lpoints_out_kernel<<<((B * S * c2) + 255) / 256, 256, 0, stream>>>(
            fout, out + olp[L + 1], B, S, c2);

        curIn ^= 1;
        featIn ^= 1;
    }
}
